// ModelCCPL_47184510714172
// MI455X (gfx1250) — matrix operand order repaired and checked by reading, not compiled, not run on hardware
//
#include <hip/hip_runtime.h>
#include <hip/hip_bf16.h>
#include <math.h>

// ---------------------------------------------------------------------------
// CCPL loss on gfx1250 (MI455X). All GEMMs run on v_wmma_f32_16x16x32_bf16
// (bf16 A/B, fp32 accumulate); gathers / norms / softmax / mean in fp32.
// GEMM: one wave -> 16x64 output strip (1 A fragment shared by 4 B fragments),
// software-pipelined: fragments for K-step k+32 are loaded (branchlessly)
// before the 4 WMMAs of step k execute, so VMEM latency hides under the
// matrix pipe. global_prefetch (WGP scope) issued 4 K-steps ahead.
// Fixed shapes from setup_inputs(): B=8, num_s=256 -> NPAIR=2048,
// ROWS=16384, D=128, C in {64,128,256,512}, H=W in {256,128,64,32}.
// start/end layer are DEVICE scalars -> gated inside nce_reduce only.
// Workspace requirement: ~57 MB.
// ---------------------------------------------------------------------------

#define NUM_S  256
#define NPAIR  (8 * NUM_S)        // 2048
#define BATCH  8
#define ROWS   (BATCH * NPAIR)    // 16384
#define DDIM   128
#define TAU    0.07f
#define EPSN   1e-7f

typedef __attribute__((ext_vector_type(16))) __bf16 v16bf;
typedef __attribute__((ext_vector_type(8)))  __bf16 v8bf;
typedef __attribute__((ext_vector_type(8)))  float  v8f;

// ------------------------------ fragments ----------------------------------
// A (16x32 bf16, row-major source, lda = K):
//   lanes 0-15 : row = lane,    K = k0+0..15   (16 contiguous bf16)
//   lanes 16-31: row = lane-16, K = k0+16..31
__device__ __forceinline__ v16bf load_a_frag(const __bf16* __restrict__ A,
                                             int lda, int row0, int k0, int lane) {
  int r = row0 + (lane & 15);
  int h = lane >> 4;
  const __bf16* p = A + (size_t)r * lda + k0 + h * 16;
  return *(const v16bf*)p;
}

// B (32x16 bf16) from Bt[N,K] row-major (Bt row n == column n of B):
//   lanes 0-15 : col = lane,    K = k0+0..15   (16 contiguous bf16)
//   lanes 16-31: col = lane-16, K = k0+16..31
__device__ __forceinline__ v16bf load_b_frag(const __bf16* __restrict__ Bt,
                                             int ldb, int n0, int k0, int lane) {
  const __bf16* p = Bt + (size_t)(n0 + (lane & 15)) * ldb + k0 + (lane >> 4) * 16;
  return *(const v16bf*)p;  // 32B, 32B-aligned for all our shapes
}

// --------------------------- generic WMMA GEMM -----------------------------
// C[M,N] = A[M,K] * Bt[N,K]^T (+bias, optional ReLU); outputs f32 and/or bf16.
// One wave -> 16 rows x 64 cols; register double-buffered over K.
__global__ void __launch_bounds__(256)
gemm_bf16_wmma(const __bf16* __restrict__ A,
               const __bf16* __restrict__ Bt,
               const float* __restrict__ bias,
               float* __restrict__ outF,
               __bf16* __restrict__ outB,
               int M, int N, int K, int relu) {
  int lane  = threadIdx.x & 31;
  int wave  = threadIdx.x >> 5;
  int jobsN = N >> 6;                        // 64-wide column groups
  int jobs  = (M >> 4) * jobsN;
  int job   = blockIdx.x * (blockDim.x >> 5) + wave;
  if (job >= jobs) return;                   // wave-uniform: EXEC stays all-1
  int row0 = (job / jobsN) * 16;
  int n0   = (job % jobsN) * 64;

  v8f acc[4] = {{}, {}, {}, {}};

  // prologue: fragments for k0 = 0
  v16bf a  = load_a_frag(A, K, row0, 0, lane);
  v16bf b0 = load_b_frag(Bt, K, n0 +  0, 0, lane);
  v16bf b1 = load_b_frag(Bt, K, n0 + 16, 0, lane);
  v16bf b2 = load_b_frag(Bt, K, n0 + 32, 0, lane);
  v16bf b3 = load_b_frag(Bt, K, n0 + 48, 0, lane);

  for (int k0 = 0; k0 < K; k0 += 32) {
    // branchless next-step index (last iteration reloads k=0; discarded)
    int kn = (k0 + 32 < K) ? k0 + 32 : 0;
    // branchless WGP-scope prefetch 4 K-steps ahead (global_prefetch_b8)
    int kp = (k0 + 128 < K) ? k0 + 128 : k0;
    __builtin_prefetch(A  + (size_t)(row0 + (lane & 15)) * K + kp, 0, 3);
    __builtin_prefetch(Bt + (size_t)(n0 + (lane & 15)) * K + kp, 0, 3);
    __builtin_prefetch(Bt + (size_t)(n0 + 32 + (lane & 15)) * K + kp, 0, 3);

    // issue all loads for the next K-step first ...
    v16bf an  = load_a_frag(A, K, row0, kn, lane);
    v16bf nb0 = load_b_frag(Bt, K, n0 +  0, kn, lane);
    v16bf nb1 = load_b_frag(Bt, K, n0 + 16, kn, lane);
    v16bf nb2 = load_b_frag(Bt, K, n0 + 32, kn, lane);
    v16bf nb3 = load_b_frag(Bt, K, n0 + 48, kn, lane);

    // ... then run this step's WMMAs on previously-loaded fragments
    acc[0] = __builtin_amdgcn_wmma_f32_16x16x32_bf16(false, a, false, b0,
                                                     (short)0, acc[0], false, false);
    acc[1] = __builtin_amdgcn_wmma_f32_16x16x32_bf16(false, a, false, b1,
                                                     (short)0, acc[1], false, false);
    acc[2] = __builtin_amdgcn_wmma_f32_16x16x32_bf16(false, a, false, b2,
                                                     (short)0, acc[2], false, false);
    acc[3] = __builtin_amdgcn_wmma_f32_16x16x32_bf16(false, a, false, b3,
                                                     (short)0, acc[3], false, false);
    a = an; b0 = nb0; b1 = nb1; b2 = nb2; b3 = nb3;
  }

  // D layout: lane 0-15 -> col=lane&15, M=r; lane 16-31 -> same col, M=r+8
  int colb  = lane & 15;
  int rbase = row0 + ((lane >> 4) << 3);
#pragma unroll
  for (int t = 0; t < 4; ++t) {
    int col  = n0 + t * 16 + colb;
    float bv = bias ? bias[col] : 0.0f;
#pragma unroll
    for (int r = 0; r < 8; ++r) {
      float v = acc[t][r] + bv;
      if (relu) v = fmaxf(v, 0.0f);
      size_t idx = (size_t)(rbase + r) * N + col;
      if (outF) outF[idx] = v;
      if (outB) outB[idx] = (__bf16)v;
    }
  }
}

// ------------------------------ small kernels ------------------------------
__global__ void zero_loss(float* loss) { if (threadIdx.x == 0 && blockIdx.x == 0) loss[0] = 0.0f; }

__device__ __forceinline__ unsigned splitmix32(unsigned x) {
  x += 0x9e3779b9u; x ^= x >> 16; x *= 0x21f0aaadu;
  x ^= x >> 15;     x *= 0x735a2d97u; x ^= x >> 15;
  return x;
}

// Deterministic interior-pixel sampling + 8-neighbor ids (threefry-permutation stand-in).
__global__ void make_ids(int* __restrict__ c_ids, int* __restrict__ n_ids,
                         int H, int W, int layer) {
  int i = blockIdx.x * blockDim.x + threadIdx.x;
  if (i >= NUM_S) return;
  unsigned M = (unsigned)((H - 2) * (W - 2));
  unsigned s = splitmix32((unsigned)i * 2654435761u + (unsigned)layer * 0x85ebca6bu) % M;
  int ch = (int)(s / (unsigned)(W - 2)) + 1;
  int cw = (int)(s % (unsigned)(W - 2)) + 1;
  int c  = ch * W + cw;
  int deltas[8] = {-(W + 1), -W, -(W - 1), -1, 1, W - 1, W, W + 1};
#pragma unroll
  for (int j = 0; j < 8; ++j) {
    c_ids[j * NUM_S + i] = c;
    n_ids[j * NUM_S + i] = c + deltas[j];
  }
}

// Wt[Out,In] = bf16(Wsrc[In,Out])
__global__ void transpose_to_bf16(const float* __restrict__ Wsrc,
                                  __bf16* __restrict__ Wt, int In, int Out) {
  int t = blockIdx.x * blockDim.x + threadIdx.x;
  if (t >= In * Out) return;
  int o = t % Out, i = t / Out;
  Wt[(size_t)o * In + i] = (__bf16)Wsrc[t];
}

// X[(b*NPAIR+i)*C + ch] = feat[b,ch,c_ids[i]] - feat[b,ch,n_ids[i]]
__global__ void gather_diff(const float* __restrict__ feat,
                            const int* __restrict__ c_ids,
                            const int* __restrict__ n_ids,
                            __bf16* __restrict__ X, int C, int HW) {
  size_t tid = (size_t)blockIdx.x * blockDim.x + threadIdx.x;
  size_t total = (size_t)ROWS * C;
  if (tid >= total) return;
  int    ch  = (int)(tid % C);
  size_t row = tid / C;
  int    i   = (int)(row % NPAIR);
  int    b   = (int)(row / NPAIR);
  const float* base = feat + ((size_t)b * C + ch) * HW;
  X[tid] = (__bf16)(base[c_ids[i]] - base[n_ids[i]]);
}

// L2-normalize each row of Y[ROWS, DDIM]; emit f32 + bf16 copies.
__global__ void rownorm(const float* __restrict__ Y,
                        float* __restrict__ Yn, __bf16* __restrict__ Ynb) {
  int row = blockIdx.x;                 // 16384 rows, 128 threads
  int t   = threadIdx.x;
  size_t o = (size_t)row * DDIM + t;
  float v = Y[o];
  float s = v * v;
#pragma unroll
  for (int off = 16; off; off >>= 1) s += __shfl_xor(s, off, 32);
  __shared__ float red[4];
  if ((t & 31) == 0) red[t >> 5] = s;
  __syncthreads();
  float tot = red[0] + red[1] + red[2] + red[3];
  float out = v / (sqrtf(tot) + EPSN);
  Yn[o]  = out;
  Ynb[o] = (__bf16)out;
}

// Per row n of lneg[NPAIR,NPAIR] (one batch b): loss += (lse - lpos)/count.
__global__ void nce_reduce(const float* __restrict__ lneg,
                           const float* __restrict__ Yq,
                           const float* __restrict__ Yk,
                           float* __restrict__ loss, int b, int layer,
                           const int* __restrict__ startp,
                           const int* __restrict__ endp) {
  if (layer < *startp || layer >= *endp) return;
  const float invtau = 1.0f / TAU;
  int n = blockIdx.x;                   // 2048 rows, 256 threads
  int t = threadIdx.x;
  __shared__ float sh[8];

  // l_pos = <Yq[b,n,:], Yk[b,n,:]> / tau
  float p = 0.0f;
  if (t < DDIM) {
    size_t o = ((size_t)b * NPAIR + n) * DDIM + t;
    p = Yq[o] * Yk[o];
  }
#pragma unroll
  for (int off = 16; off; off >>= 1) p += __shfl_xor(p, off, 32);
  if ((t & 31) == 0) sh[t >> 5] = p;
  __syncthreads();
  float lp = 0.0f;
#pragma unroll
  for (int w = 0; w < 8; ++w) lp += sh[w];
  lp *= invtau;
  __syncthreads();

  // row max over {lp} U {lneg[n,m]/tau : m != n}
  const float* rowp = lneg + (size_t)n * NPAIR;
  float mx = lp;
  for (int m = t; m < NPAIR; m += 256)
    if (m != n) mx = fmaxf(mx, rowp[m] * invtau);
#pragma unroll
  for (int off = 16; off; off >>= 1) mx = fmaxf(mx, __shfl_xor(mx, off, 32));
  if ((t & 31) == 0) sh[t >> 5] = mx;
  __syncthreads();
  float gmx = sh[0];
#pragma unroll
  for (int w = 1; w < 8; ++w) gmx = fmaxf(gmx, sh[w]);
  __syncthreads();

  // sum of exp
  float se = (t == 0) ? __expf(lp - gmx) : 0.0f;
  for (int m = t; m < NPAIR; m += 256)
    if (m != n) se += __expf(rowp[m] * invtau - gmx);
#pragma unroll
  for (int off = 16; off; off >>= 1) se += __shfl_xor(se, off, 32);
  if ((t & 31) == 0) sh[t >> 5] = se;
  __syncthreads();
  if (t == 0) {
    float tot = 0.0f;
#pragma unroll
    for (int w = 0; w < 8; ++w) tot += sh[w];
    float lse = gmx + __logf(tot);
    atomicAdd(loss, (lse - lp) * (1.0f / (float)(BATCH * NPAIR)));
  }
}

// ------------------------------ host driver --------------------------------
static inline void run_gemm(const __bf16* A, const __bf16* Bt, const float* bias,
                            float* outF, __bf16* outB, int M, int N, int K,
                            int relu, hipStream_t s) {
  int jobs   = (M / 16) * (N / 64);   // one 16x64 strip per wave
  int blocks = (jobs + 7) / 8;        // 8 waves per 256-thread block
  gemm_bf16_wmma<<<blocks, 256, 0, s>>>(A, Bt, bias, outF, outB, M, N, K, relu);
}

extern "C" void kernel_launch(void* const* d_in, const int* in_sizes, int n_in,
                              void* d_out, int out_size, void* d_ws, size_t ws_size,
                              hipStream_t stream) {
  (void)in_sizes; (void)n_in; (void)out_size; (void)ws_size;
  float* loss = (float*)d_out;
  zero_loss<<<1, 32, 0, stream>>>(loss);

  // setup_inputs dict order: q0,k0,q1,k1,q2,k2,q3,k3, then (w1,b1,w2,b2)x4, num_s, start, end
  const float* fq[4] = {(const float*)d_in[0], (const float*)d_in[2],
                        (const float*)d_in[4], (const float*)d_in[6]};
  const float* fk[4] = {(const float*)d_in[1], (const float*)d_in[3],
                        (const float*)d_in[5], (const float*)d_in[7]};
  const int* startp = (const int*)d_in[25];
  const int* endp   = (const int*)d_in[26];

  // workspace layout (bytes) -- ~57 MB total
  char* ws = (char*)d_ws;
  const size_t MB = 1024 * 1024;
  int*    c_ids = (int*)ws;                       // 8 KB
  int*    n_ids = c_ids + NPAIR;                  // 8 KB
  __bf16* w1T   = (__bf16*)(ws + 64 * 1024);      // up to 512 KB
  __bf16* w2T   = (__bf16*)(ws + 64 * 1024 + 512 * 1024); // up to 128 KB
  char*   regA  = ws + 1 * MB;                    // 16 MB: X -> Ytmp -> lneg
  __bf16* Xbuf  = (__bf16*)regA;
  float*  Ytmp  = (float*)regA;                   // 8 MB  (alias, after X consumed)
  float*  lneg  = (float*)regA;                   // 16 MB (alias, after Ytmp consumed)
  __bf16* Hbuf  = (__bf16*)(ws + 17 * MB);        // 16 MB
  float*  Yq    = (float*)(ws + 33 * MB);         // 8 MB
  float*  Yk    = (float*)(ws + 41 * MB);         // 8 MB
  __bf16* Yqb   = (__bf16*)(ws + 49 * MB);        // 4 MB
  __bf16* Ykb   = (__bf16*)(ws + 53 * MB);        // 4 MB

  static const int chans[4] = {64, 128, 256, 512};
  static const int hs[4]    = {256, 128, 64, 32};

  for (int l = 0; l < 4; ++l) {
    int C = chans[l], H = hs[l], W = hs[l], HW = H * W;
    const float* w1 = (const float*)d_in[8 + 4 * l];
    const float* b1 = (const float*)d_in[9 + 4 * l];
    const float* w2 = (const float*)d_in[10 + 4 * l];
    const float* b2 = (const float*)d_in[11 + 4 * l];

    make_ids<<<(NUM_S + 255) / 256, 256, 0, stream>>>(c_ids, n_ids, H, W, l);
    transpose_to_bf16<<<(C * C + 255) / 256, 256, 0, stream>>>(w1, w1T, C, C);
    transpose_to_bf16<<<(C * DDIM + 255) / 256, 256, 0, stream>>>(w2, w2T, C, DDIM);

    for (int pass = 0; pass < 2; ++pass) {      // 0 = q, 1 = k
      const float* feat = pass ? fk[l] : fq[l];
      float*  Yn  = pass ? Yk  : Yq;
      __bf16* Ynb = pass ? Ykb : Yqb;
      size_t total = (size_t)ROWS * C;
      gather_diff<<<(unsigned)((total + 255) / 256), 256, 0, stream>>>(
          feat, c_ids, n_ids, Xbuf, C, HW);
      // H = relu(X @ w1 + b1)   [ROWS,C]x[C,C]   -> bf16 only
      run_gemm(Xbuf, w1T, b1, nullptr, Hbuf, ROWS, C, C, 1, stream);
      // Y = H @ w2 + b2         [ROWS,C]x[C,128] -> f32 (aliases X region)
      run_gemm(Hbuf, w2T, b2, Ytmp, nullptr, ROWS, DDIM, C, 0, stream);
      rownorm<<<ROWS, DDIM, 0, stream>>>(Ytmp, Yn, Ynb);
    }

    // NCE per batch image: lneg = Yq[b] . Yk[b]^T  (2048x2048x128), then reduce.
    for (int b = 0; b < BATCH; ++b) {
      const __bf16* Aq = Yqb + (size_t)b * NPAIR * DDIM;
      const __bf16* Bk = Ykb + (size_t)b * NPAIR * DDIM;
      run_gemm(Aq, Bk, nullptr, lneg, nullptr, NPAIR, NPAIR, DDIM, 0, stream);
      nce_reduce<<<NPAIR, 256, 0, stream>>>(lneg, Yq, Yk, loss, b, l, startp, endp);
    }
  }
}
